// LinearRNN_46677704573036
// MI455X (gfx1250) — matrix operand order repaired and checked by reading, not compiled, not run on hardware
//
#include <hip/hip_runtime.h>

// Linear RNN h_t = B h_{t-1} + U*u_t on gfx1250 via chunked-scan -> bf16 WMMA GEMMs.
// kH=2048 hidden, kT=8192 steps, chunks: kC=128 chunks of kS=64 steps.

static constexpr int kH = 2048;
static constexpr int kT = 8192;
static constexpr int kS = 64;    // chunk length
static constexpr int kC = 128;   // number of chunks (kT / kS)

typedef __attribute__((ext_vector_type(16))) __bf16 v16bf;
typedef __attribute__((ext_vector_type(8)))  float  v8f;

union FragAB { unsigned u[8]; v16bf v; };

__device__ __forceinline__ unsigned short f2bf(float f) {
    unsigned u = __builtin_bit_cast(unsigned, f);
    u += 0x7FFFu + ((u >> 16) & 1u);            // round-to-nearest-even
    return (unsigned short)(u >> 16);
}
__device__ __forceinline__ float bf2f(unsigned short s) {
    return __builtin_bit_cast(float, (unsigned)s << 16);
}

// ---------------------------------------------------------------------------
// Generic bf16 GEMM with f32 accumulate:
//   D[M x N] = A_rm[M x K] @ Bop  (+ Uvec[m] * u_s[n * u_stride])
// A_rm : row-major bf16.  B_cm : operand columns, B_cm[n*K + k] (bf16).
// Outputs (all optional): outF  (f32, column-major, col stride outF_cs)
//                         outCM (bf16, column-major, col stride outCM_cs)
//                         outRM (bf16, row-major, row stride outRM_rs)
// Block = 256 threads (8 wave32), each wave owns a 64x16 strip (4 WMMA tiles).
// Grid  = (M/512, N/16).
// ---------------------------------------------------------------------------
__global__ __launch_bounds__(256) void rnn_gemm_bf16(
    const unsigned short* __restrict__ A_rm,
    const unsigned short* __restrict__ B_cm,
    int M, int N, int K,
    float* __restrict__ outF, long long outF_cs,
    unsigned short* __restrict__ outCM, int outCM_cs,
    unsigned short* __restrict__ outRM, int outRM_rs,
    const float* __restrict__ Uvec, const float* __restrict__ u_s, int u_stride)
{
    __shared__ unsigned ldsX[16 * 16];   // 32(K) x 16(N) bf16 B-operand tile

    const int tid  = threadIdx.x;
    const int wave = tid >> 5;
    const int lane = tid & 31;
    const int hlf  = lane >> 4;          // lane half (0/1)
    const int nl   = lane & 15;

    const int n0    = blockIdx.y * 16;
    const int mbase = blockIdx.x * 512 + wave * 64;

    v8f acc[4];
    #pragma unroll
    for (int t = 0; t < 4; ++t)
        #pragma unroll
        for (int r = 0; r < 8; ++r) acc[t][r] = 0.0f;

    // cooperative staging coordinates: one u32 (K-pair) per thread
    const int js = tid >> 4;             // operand column 0..15
    const int ps = tid & 15;             // K-pair index 0..15
    const unsigned short* bcol = B_cm + (size_t)(n0 + js) * K + 2 * ps;

    for (int k0 = 0; k0 < K; k0 += 32) {
        ldsX[js * 16 + ps] = *(const unsigned*)(bcol + k0);
        __syncthreads();

        // B-operand frag: lane holds column nl, K = hlf*16 + 2v .. +1
        FragAB fb;
        #pragma unroll
        for (int v = 0; v < 8; ++v)
            fb.u[v] = ldsX[nl * 16 + hlf * 8 + v];

        #pragma unroll
        for (int t = 0; t < 4; ++t) {
            const int m = mbase + t * 16 + nl;
            const unsigned short* arow = A_rm + (size_t)m * K + k0;
            // A frag: K = hlf*16 + 2v .. +1 (matches B frag)
            FragAB fa;
            #pragma unroll
            for (int v = 0; v < 8; ++v) {
                const int k = hlf * 16 + v * 2;
                fa.u[v] = *(const unsigned*)(arow + k);
            }
            acc[t] = __builtin_amdgcn_wmma_f32_16x16x32_bf16(
                false, fa.v, false, fb.v, (short)0, acc[t], false, false);
        }
        __syncthreads();
        if (k0 + 32 < K)
            __builtin_prefetch(bcol + k0 + 32, 0, 1);   // global_prefetch_b8
    }

    const int n_g = n0 + nl;
    float uval = 0.0f;
    if (Uvec) uval = u_s[(size_t)n_g * u_stride];

    #pragma unroll
    for (int t = 0; t < 4; ++t) {
        const int m0 = mbase + t * 16 + hlf * 8;     // 8 consecutive rows
        float vals[8];
        #pragma unroll
        for (int r = 0; r < 8; ++r) vals[r] = acc[t][r];
        if (Uvec) {
            #pragma unroll
            for (int r = 0; r < 8; ++r) vals[r] += Uvec[m0 + r] * uval;
        }
        if (outF) {
            float* pf = outF + (size_t)n_g * outF_cs + m0;
            *(float4*)(pf)     = make_float4(vals[0], vals[1], vals[2], vals[3]);
            *(float4*)(pf + 4) = make_float4(vals[4], vals[5], vals[6], vals[7]);
        }
        if (outCM) {
            union { unsigned short h[8]; uint4 q; } pk;
            #pragma unroll
            for (int r = 0; r < 8; ++r) pk.h[r] = f2bf(vals[r]);
            *(uint4*)(outCM + (size_t)n_g * outCM_cs + m0) = pk.q;
        }
        if (outRM) {
            #pragma unroll
            for (int r = 0; r < 8; ++r)
                outRM[(size_t)(m0 + r) * outRM_rs + n_g] = f2bf(vals[r]);
        }
    }
}

// Inter-chunk carry: prev_out = B^64 @ prev_in + e_in ; also emit bf16 column.
__global__ __launch_bounds__(256) void rnn_carry_gemv(
    const unsigned short* __restrict__ Bs_rm,
    const float* __restrict__ prev_in,
    const float* __restrict__ e_in,
    float* __restrict__ prev_out,
    unsigned short* __restrict__ xcol_out)
{
    const int m = blockIdx.x * 256 + threadIdx.x;
    const unsigned* row = (const unsigned*)(Bs_rm + (size_t)m * kH);
    float acc = e_in[m];
    #pragma unroll 8
    for (int kk = 0; kk < kH / 2; ++kk) {
        const unsigned pr = row[kk];
        acc += bf2f((unsigned short)(pr & 0xFFFFu)) * prev_in[2 * kk];
        acc += bf2f((unsigned short)(pr >> 16))     * prev_in[2 * kk + 1];
    }
    prev_out[m] = acc;
    xcol_out[m] = f2bf(acc);
}

// Convert B (f32) into row-major and column-major bf16 copies.
__global__ __launch_bounds__(256) void rnn_convB(
    const float* __restrict__ B, unsigned short* __restrict__ rm,
    unsigned short* __restrict__ cm)
{
    const int i = blockIdx.x * 256 + threadIdx.x;       // kH*kH threads
    const float v = B[i];
    rm[i] = f2bf(v);
    const int r = i >> 11, c = i & (kH - 1);
    cm[(size_t)c * kH + r] = f2bf(v);
}

__global__ __launch_bounds__(256) void rnn_zero_init(
    unsigned short* __restrict__ X0, unsigned short* __restrict__ X1,
    float* __restrict__ P)
{
    const int i = blockIdx.x * 256 + threadIdx.x;       // kC*kH threads
    X0[i] = 0;
    if (i < kH) { X1[i] = 0; P[i] = 0.0f; }
}

__global__ __launch_bounds__(256) void rnn_zero_col0(
    unsigned short* __restrict__ X1, float* __restrict__ P)
{
    const int i = blockIdx.x * 256 + threadIdx.x;       // kH threads
    X1[i] = 0; P[i] = 0.0f;
}

// y_t = C_w . h_t  (one block per t, LDS tree reduction)
__global__ __launch_bounds__(256) void rnn_yproj(
    const float* __restrict__ Cw, const float* __restrict__ hist,
    float* __restrict__ y)
{
    __shared__ float red[256];
    const int t = blockIdx.x;
    float a = 0.0f;
    for (int i = threadIdx.x; i < kH; i += 256)
        a += Cw[i] * hist[(size_t)t * kH + i];
    red[threadIdx.x] = a;
    __syncthreads();
    for (int s = 128; s > 0; s >>= 1) {
        if (threadIdx.x < s) red[threadIdx.x] += red[threadIdx.x + s];
        __syncthreads();
    }
    if (threadIdx.x == 0) y[t] = red[0];
}

extern "C" void kernel_launch(void* const* d_in, const int* in_sizes, int n_in,
                              void* d_out, int out_size, void* d_ws, size_t ws_size,
                              hipStream_t stream) {
    (void)in_sizes; (void)n_in; (void)out_size; (void)ws_size;
    const float* u_seq = (const float*)d_in[0];
    const float* Bf    = (const float*)d_in[1];
    const float* Uvec  = (const float*)d_in[2];
    const float* Cw    = (const float*)d_in[3];

    float* y    = (float*)d_out;          // [kT]
    float* hist = (float*)d_out + kT;     // [kT][kH] row-major

    char* ws = (char*)d_ws;
    const size_t MATB = (size_t)kH * kH * sizeof(unsigned short);   // 8 MB
    unsigned short* Brm  = (unsigned short*)(ws + 0 * MATB);
    unsigned short* Bcm  = (unsigned short*)(ws + 1 * MATB);
    unsigned short* t0rm = (unsigned short*)(ws + 2 * MATB);
    unsigned short* t0cm = (unsigned short*)(ws + 3 * MATB);
    unsigned short* t1rm = (unsigned short*)(ws + 4 * MATB);
    unsigned short* t1cm = (unsigned short*)(ws + 5 * MATB);
    char* p = ws + 6 * MATB;
    unsigned short* X0 = (unsigned short*)p; p += (size_t)kC * kH * 2;
    unsigned short* X1 = (unsigned short*)p; p += (size_t)kC * kH * 2;
    float* Ebuf = (float*)p; p += (size_t)kC * kH * 4;   // chunk-local endpoints
    float* Pbuf = (float*)p;                              // carries (f32)

    // 1) B -> bf16 (row-major + column-major)
    rnn_convB<<<dim3((kH * kH) / 256), dim3(256), 0, stream>>>(Bf, Brm, Bcm);
    // 2) zero pass-1 initial state, carry col 0
    rnn_zero_init<<<dim3((kC * kH) / 256), dim3(256), 0, stream>>>(X0, X1, Pbuf);

    // 3) B^64 via 6 repeated squarings (ping-pong), full 2048^3 WMMA GEMMs
    const unsigned short* cr = Brm; const unsigned short* cc = Bcm;
    for (int i = 0; i < 6; ++i) {
        unsigned short* drm = (i & 1) ? t1rm : t0rm;
        unsigned short* dcm = (i & 1) ? t1cm : t0cm;
        rnn_gemm_bf16<<<dim3(kH / 512, kH / 16), dim3(256), 0, stream>>>(
            cr, cc, kH, kH, kH, nullptr, 0, dcm, kH, drm, kH,
            nullptr, nullptr, 0);
        cr = drm; cc = dcm;
    }
    const unsigned short* Bpow = cr;      // B^64 row-major (bf16)

    // 4) pass 1: chunk-local recursion H_s = B@H_{s-1} + V_s from zero state
    for (int s = 0; s < kS; ++s) {
        const unsigned short* xin = (s & 1) ? X1 : X0;
        unsigned short*       xout = (s & 1) ? X0 : X1;
        float* ef = (s == kS - 1) ? Ebuf : nullptr;     // capture endpoints
        rnn_gemm_bf16<<<dim3(kH / 512, kC / 16), dim3(256), 0, stream>>>(
            Brm, xin, kH, kC, kH, ef, kH, xout, kH, nullptr, 0,
            Uvec, u_seq + s, kS);
    }

    // 5) reset carry column 0 (chunk 0 starts from h = 0)
    rnn_zero_col0<<<dim3(kH / 256), dim3(256), 0, stream>>>(X1, Pbuf);

    // 6) sequential inter-chunk carry scan: prev_c = B^64 prev_{c-1} + e_{c-1}
    for (int c = 1; c < kC; ++c) {
        rnn_carry_gemv<<<dim3(kH / 256), dim3(256), 0, stream>>>(
            Bpow, Pbuf + (size_t)(c - 1) * kH, Ebuf + (size_t)(c - 1) * kH,
            Pbuf + (size_t)c * kH, X1 + (size_t)c * kH);
    }

    // 7) pass 2: recursion seeded with carries; stream h_hist slices
    //    element (m, chunk c) at step s -> hist[(c*kS + s)*kH + m]
    for (int s = 0; s < kS; ++s) {
        const unsigned short* xin = (s & 1) ? X0 : X1;
        unsigned short*       xout = (s & 1) ? X1 : X0;
        rnn_gemm_bf16<<<dim3(kH / 512, kC / 16), dim3(256), 0, stream>>>(
            Brm, xin, kH, kC, kH,
            hist + (size_t)s * kH, (long long)kS * kH,
            xout, kH, nullptr, 0, Uvec, u_seq + s, kS);
    }

    // 8) y = h_hist @ C_w^T
    rnn_yproj<<<dim3(kT), dim3(256), 0, stream>>>(Cw, hist, y);
}
